// CNN_noninvariant_28226525069675
// MI455X (gfx1250) — matrix operand order repaired and checked by reading, not compiled, not run on hardware
//
#include <hip/hip_runtime.h>
#include <cstdint>

typedef __attribute__((ext_vector_type(16))) _Float16 v16h;
typedef __attribute__((ext_vector_type(8)))  float    v8f;

#define L2C    65536
#define NTOT   131072
#define FINC   8
#define FOUTC  8
#define KTAP   13
#define CDIM   104          // FIN*K
#define TILE   16           // edges per wave/WMMA tile
#define WAVES  4
#define TPB    128

// LDS: per-edge mask row padded 832 -> 848 B (212 dwords == 20 banks mod 64,
// so the 16 edge-lanes of a b64 mask read land on 16 distinct bank pairs).
#define EDGE_STR 848
#define MASK_STR (TILE*EDGE_STR + 64)      // 13632 (+64 pad for tail b64 reads)

// dynamic LDS layout (bytes)
#define OFF_W    0                         // 1024 dwords swizzled f16 weights (both branches)
#define OFF_KER  4096                      // 4 waves * 208 ints  = 3328
#define OFF_Z    7424                      // 4 waves * 128 floats = 2048
#define OFF_MASK 9472
#define SMEM_SZ  (OFF_MASK + WAVES*MASK_STR)   // 64000 B

__device__ __forceinline__ uint32_t pkf16(float a, float b) {
    _Float16 ha = (_Float16)a, hb = (_Float16)b;
    uint16_t ua = __builtin_bit_cast(uint16_t, ha);
    uint16_t ub = __builtin_bit_cast(uint16_t, hb);
    return (uint32_t)ua | ((uint32_t)ub << 16);
}

// gather one x value for flattened channel c (c = i*13 + k); zero-pad c >= 104
__device__ __forceinline__ float gx(const float* __restrict__ x, const int* kerrow, int c) {
    if (c >= CDIM) return 0.0f;
    int i = (c * 5042) >> 16;       // floor(c/13), exact for c < 6553
    int k = c - 13 * i;
    int idx = kerrow[k];            // LDS read
    return x[(size_t)i * NTOT + idx];
}

__global__ __launch_bounds__(TPB)
void edge_masked_conv_kernel(const float* __restrict__ x,
                             const float* __restrict__ Wh, const float* __restrict__ Wv,
                             const float* __restrict__ bh, const float* __restrict__ bv,
                             const uint8_t* __restrict__ mh, const uint8_t* __restrict__ mv,
                             const int* __restrict__ kh, const int* __restrict__ kvr,
                             const int* __restrict__ eh, const int* __restrict__ ev,
                             float* __restrict__ out)
{
    extern __shared__ char smem[];
    uint32_t* wlds = (uint32_t*)(smem + OFF_W);

    const int tid  = threadIdx.x;
    const int lane = tid & 31;
    const int wv   = tid >> 5;

    // ---- one-time per block: swizzle W (both branches) into f16 B-fragment layout ----
    // B layout (32x16 f16): lane half hh: VGPR r -> K = 8*hh + 2r (+8 if r>=4) (same as A).
    for (int n = tid; n < 1024; n += TPB) {
        int r = n & 7, hh = (n >> 3) & 1, q = (n >> 4) & 3, o = (n >> 6) & 7, br = (n >> 9) & 1;
        int c0 = 32 * q + 8 * hh + 2 * r + ((r & 4) ? 8 : 0);
        const float* Wp = br ? Wv : Wh;
        float w0 = (c0     < CDIM) ? Wp[o * CDIM + c0]     : 0.0f;
        float w1 = (c0 + 1 < CDIM) ? Wp[o * CDIM + c0 + 1] : 0.0f;
        wlds[n] = pkf16(w0, w1);
    }
    __syncthreads();

    // ---- tile assignment: one wave = 16 edges of one branch ----
    const int wgid   = blockIdx.x * WAVES + wv;   // 0..8191
    const int br     = wgid >> 12;                // 0 = hor, 1 = vert
    const int tileid = wgid & 4095;
    const int ebase  = tileid * TILE;

    const uint8_t* maskp = br ? mv  : mh;
    const int*     kerp  = br ? kvr : kh;
    const float*   biasp = br ? bv  : bh;
    const int*     elist = br ? ev  : eh;

    uint8_t* maskw = (uint8_t*)(smem + OFF_MASK) + wv * MASK_STR;
    int*     kerw  = (int*)(smem + OFF_KER) + wv * (TILE * KTAP);
    float*   zw    = (float*)(smem + OFF_Z) + wv * (FOUTC * TILE);

    // ---- 1) async copy this tile's masks (16*832 B) into LDS, de-linearized into
    //         848 B-padded rows (16 B chunks never cross an 832 B row: 832 = 52*16) ----
    {
        unsigned long long gp = (unsigned long long)(uintptr_t)(maskp + (size_t)ebase * (FOUTC * CDIM));
        uint32_t lbase = (uint32_t)(uintptr_t)maskw;
        uint32_t g = (uint32_t)lane * 16u;
        #pragma unroll
        for (int it = 0; it < 26; ++it, g += 512u) {
            uint32_t row   = ((g >> 6) * 5042u) >> 16;   // g / 832
            uint32_t laddr = lbase + g + (row << 4);     // + 16 B pad per preceding row
            asm volatile("global_load_async_to_lds_b128 %0, %1, %2"
                         :: "v"(laddr), "v"(g), "s"(gp) : "memory");
        }
    }

    // ---- 2) ker rows for this tile -> LDS ----
    for (int t = lane; t < TILE * KTAP; t += 32) kerw[t] = kerp[ebase * KTAP + t];
    asm volatile("s_wait_dscnt 0" ::: "memory");

    // ---- 3) gather x into per-lane A-fragment f16 pairs (shared across all 8 outputs) ----
    // A layout (16x32 f16): lane half hh: VGPR r -> K = 8*hh + 2r (+8 if r>=4); row = lane&15.
    const int e16 = lane & 15;
    const int hh  = lane >> 4;
    const int* kerrow = kerw + e16 * KTAP;
    uint32_t xp[32];
    #pragma unroll
    for (int t = 0; t < 32; ++t) {
        int q  = t >> 3, r = t & 7;
        int c0 = 32 * q + 8 * hh + 2 * r + ((r & 4) ? 8 : 0);
        xp[t] = pkf16(gx(x, kerrow, c0), gx(x, kerrow, c0 + 1));
    }

    asm volatile("s_wait_asynccnt 0" ::: "memory");

    // ---- 4) per output channel: mask-AND A, broadcast-B, K=128 WMMA chain ----
    #pragma unroll
    for (int o = 0; o < FOUTC; ++o) {
        v8f acc = {};
        const uint32_t eoff = (uint32_t)e16 * EDGE_STR + (uint32_t)o * CDIM;
        #pragma unroll
        for (int q = 0; q < 4; ++q) {
            union { uint32_t u[8]; uint4 v[2]; v16h h; } af, bf;
            const uint4* wp4 = (const uint4*)(wlds + ((((br * FOUTC + o) * 4 + q) * 2 + hh) * 8));
            bf.v[0] = wp4[0];
            bf.v[1] = wp4[1];

            // two aligned b64 mask reads cover this lane-half's 16 channels of chunk q;
            // keep everything in dwords so bit extracts fold into single v_bfe_i32 ops
            const uint32_t base = eoff + 32u * q + 8u * hh;
            uint64_t m0 = *(const uint64_t*)(maskw + base);
            uint64_t m1 = *(const uint64_t*)(maskw + base + 16u);
            uint32_t dd[4] = { (uint32_t)m0, (uint32_t)(m0 >> 32),
                               (uint32_t)m1, (uint32_t)(m1 >> 32) };
            #pragma unroll
            for (int r = 0; r < 8; ++r) {
                uint32_t dw = dd[r >> 1];
                int sh = (r & 1) * 16;
                uint32_t lo = (uint32_t)((int)(dw << (31 - sh)) >> 31);       // bit sh   -> 0/-1
                uint32_t hi = (uint32_t)((int)(dw << (23 - sh)) >> 31);       // bit sh+8 -> 0/-1
                uint32_t am = (lo & 0x0000FFFFu) | (hi & 0xFFFF0000u);        // -> bitop3
                af.u[r] = xp[q * 8 + r] & am;                                 // masked x
            }
            acc = __builtin_amdgcn_wmma_f32_16x16x32_f16(
                      false, af.h, false, bf.h, (short)0, acc, false, false);
        }
        // all 16 D columns are identical; column N=0 lives in lanes 0 (rows 0-7) and 16 (rows 8-15)
        if (e16 == 0) {
            float4* zp = (float4*)(zw + o * TILE + hh * 8);
            float4 a0 = { acc[0], acc[1], acc[2], acc[3] };
            float4 a1 = { acc[4], acc[5], acc[6], acc[7] };
            zp[0] = a0;
            zp[1] = a1;
        }
    }

    asm volatile("s_wait_dscnt 0" ::: "memory");

    // ---- 5) bias + (sigmoid-0.5)*SCALE epilogue; lanes 0-15 -> o 0..3, lanes 16-31 -> o 4..7 ----
    constexpr float SCALEF = (float)((2.0 + 2.0 * 2.718281828459045) / (2.718281828459045 - 1.0));
    const int col = elist[ebase + e16];
    const int ob  = hh * 4;
    #pragma unroll
    for (int oo = 0; oo < 4; ++oo) {
        int   o = ob + oo;
        float z = zw[o * TILE + e16] + biasp[o];
        float s = 1.0f / (1.0f + __expf(-z));
        out[(size_t)o * NTOT + col] = (s - 0.5f) * SCALEF;
    }
}

extern "C" void kernel_launch(void* const* d_in, const int* in_sizes, int n_in,
                              void* d_out, int out_size, void* d_ws, size_t ws_size,
                              hipStream_t stream) {
    const float*   x  = (const float*)d_in[0];
    const float*   Wh = (const float*)d_in[1];
    const float*   Wv = (const float*)d_in[2];
    const float*   bh = (const float*)d_in[3];
    const float*   bv = (const float*)d_in[4];
    const uint8_t* mh = (const uint8_t*)d_in[5];   // bool mask -> 1 byte/elem
    const uint8_t* mv = (const uint8_t*)d_in[6];
    const int*     kh = (const int*)d_in[7];
    const int*     kv = (const int*)d_in[8];
    const int*     eh = (const int*)d_in[9];
    const int*     ev = (const int*)d_in[10];
    (void)in_sizes; (void)n_in; (void)out_size; (void)d_ws; (void)ws_size;

    dim3 grid(2 * (L2C / TILE) / WAVES);   // 8192 wave-tiles / 4 waves per block = 2048 blocks
    dim3 block(TPB);
    edge_masked_conv_kernel<<<grid, block, SMEM_SZ, stream>>>(
        x, Wh, Wv, bh, bv, mh, mv, kh, kv, eh, ev, (float*)d_out);
}
